// IntraAgg_64020782514706
// MI455X (gfx1250) — matrix operand order repaired and checked by reading, not compiled, not run on hardware
//
#include <hip/hip_runtime.h>

typedef __attribute__((ext_vector_type(16))) __bf16 v16bf;
typedef __attribute__((ext_vector_type(8)))  __bf16 v8bf;
typedef __attribute__((ext_vector_type(4)))  __bf16 v4bf;
typedef __attribute__((ext_vector_type(8)))  float  v8f;
typedef __attribute__((ext_vector_type(4)))  float  v4f;

#define B_ROWS   16384
#define K_NEIGH  32
#define FEAT     128
#define EMB      128
#define MASK_THRESH 0.25f

// ---------------------------------------------------------------------------
// Kernel 0: pack an f32 row-major [K, N] weight into bf16 WMMA B-fragments.
// Fragment (kt, nt) covers K rows [kt*32, kt*32+32), N cols [nt*16, nt*16+16).
// Per ISA B layout (32x16 bf16): lanes 0-15 hold K=0..7,16..23, lanes 16-31 hold
// K=8..15,24..31; N = lane & 15; elements 8..15 are offset by 16 in K.
// Stored contiguously: dst[((kt*(N/16)+nt)*32 + lane)*16 + e]
// ---------------------------------------------------------------------------
__global__ void pack_b_kernel(const float* __restrict__ src,
                              __bf16* __restrict__ dst, int K, int N) {
    int t = blockIdx.x * blockDim.x + threadIdx.x;   // one thread per (frag,lane)
    int ntiles = N >> 4;
    int total = (K >> 5) * ntiles * 32;
    if (t >= total) return;
    int lane = t & 31;
    int frag = t >> 5;
    int kt = frag / ntiles;
    int nt = frag - kt * ntiles;
    int n  = nt * 16 + (lane & 15);
    int kbase = kt * 32 + (lane >> 4) * 8;
    __bf16* out = dst + (size_t)t * 16;
#pragma unroll
    for (int e = 0; e < 16; ++e)
        out[e] = (__bf16)src[(size_t)(kbase + e + (e >= 8 ? 8 : 0)) * N + n];
}

// ---------------------------------------------------------------------------
// Kernel 1: ragged gather + masked mean. One WAVE per batch row, each lane
// owns 4 contiguous feature columns -> every neighbor row is fetched with a
// single wave-wide global_load_b128 (32 lanes x 16 B = the full 512-B row).
//  * mask bits -> ballot_w32: denom = s_popcount of one SGPR, no butterfly
//  * neighbor index -> v_readlane into SGPR: gather uses scalar-base (GVS)
//    addressing and the zero-weight skip is a pure scalar branch.
// No LDS / ds_bpermute traffic in the bandwidth-critical loop.
// ---------------------------------------------------------------------------
__global__ __launch_bounds__(256) void agg_kernel(
    const float* __restrict__ features, const float* __restrict__ neigh_mask,
    const int* __restrict__ nodes, const int* __restrict__ neigh_idx,
    __bf16* __restrict__ selfB, __bf16* __restrict__ toB) {
    const int wave = threadIdx.x >> 5;
    const int lane = threadIdx.x & 31;
    const int b    = blockIdx.x * 8 + wave;          // one row per wave

    bool pred = neigh_mask[(size_t)b * K_NEIGH + lane] > MASK_THRESH;
    int  idx  = neigh_idx[(size_t)b * K_NEIGH + lane];

    unsigned mbits = __builtin_amdgcn_ballot_w32(pred);   // uniform SGPR mask
    float denom = (float)__builtin_popcount(mbits);
    float inv = 1.0f / fmaxf(denom, 1.0f);

    v4f acc = {0.0f, 0.0f, 0.0f, 0.0f};
#pragma unroll
    for (int k = 0; k < K_NEIGH; ++k) {
        if (mbits & (1u << k)) {                     // scalar branch
            int ik = __builtin_amdgcn_readlane(idx, k);   // SGPR index
            v4f f = *(const v4f*)(features + (size_t)ik * FEAT + lane * 4);
            acc = acc + f;
        }
    }
    acc = acc * inv;

    v4f self4 = *(const v4f*)(features + (size_t)nodes[b] * FEAT + lane * 4);

    v4bf sb, tb;
    sb.x = (__bf16)self4.x; sb.y = (__bf16)self4.y;
    sb.z = (__bf16)self4.z; sb.w = (__bf16)self4.w;
    tb.x = (__bf16)acc.x;   tb.y = (__bf16)acc.y;
    tb.z = (__bf16)acc.z;   tb.w = (__bf16)acc.w;
    *(v4bf*)(selfB + (size_t)b * FEAT + lane * 4) = sb;
    *(v4bf*)(toB   + (size_t)b * FEAT + lane * 4) = tb;
}

// ---------------------------------------------------------------------------
// Kernel 2: fused WMMA GEMMs. Block = 256 threads = 8 waves, one 16-row tile
// of the batch per block; wave w owns the 16-column tile n0 = w*16.
//   sim = relu(self @ S)            4 x v_wmma_f32_16x16x32_bf16 per wave
//   out = relu(self@W0 + to@W1 + sim@W2)   12 WMMAs per wave
// ---------------------------------------------------------------------------
union AFrag { v16bf v; v8bf h[2]; };

__device__ inline v16bf load_a_frag(const __bf16* row, int kt, int khalf) {
    // A 16x32 bf16 layout: elems 0..7  -> K = kt*32 + khalf + 0..7
    //                      elems 8..15 -> K = kt*32 + 16 + khalf + 0..7
    AFrag f;
    f.h[0] = *(const v8bf*)(row + kt * 32 + khalf);
    f.h[1] = *(const v8bf*)(row + kt * 32 + 16 + khalf);
    return f.v;
}

__device__ inline v16bf load_b_frag(const __bf16* pack, int kt, int nt, int lane) {
    AFrag f;
    const __bf16* p = pack + ((size_t)((kt * 8 + nt) * 32 + lane)) * 16;
    f.h[0] = *(const v8bf*)(p);
    f.h[1] = *(const v8bf*)(p + 8);
    return f.v;
}

__global__ __launch_bounds__(256) void fused_gemm_kernel(
    const __bf16* __restrict__ selfB, const __bf16* __restrict__ toB,
    const __bf16* __restrict__ packS, const __bf16* __restrict__ packW,
    float* __restrict__ out) {
    const int wave  = threadIdx.x >> 5;          // n-tile 0..7
    const int lane  = threadIdx.x & 31;
    const int m     = lane & 15;                 // A row within tile
    const int khalf = (lane >> 4) * 8;           // A K-subgroup select
    const int chalf = (lane >> 4) * 8;           // C/D row offset
    const size_t row0 = (size_t)blockIdx.x * 16;

    __shared__ __bf16 simLds[16 * 136];          // 272B row stride: 16B-aligned,
                                                 // +4 banks per row (no conflicts)

    const __bf16* selfRow = selfB + (row0 + m) * FEAT;
    const __bf16* toRow   = toB   + (row0 + m) * FEAT;

    v16bf a_self[4], a_to[4];
#pragma unroll
    for (int kt = 0; kt < 4; ++kt) {
        a_self[kt] = load_a_frag(selfRow, kt, khalf);
        a_to[kt]   = load_a_frag(toRow,   kt, khalf);
    }

    // ---- sim = relu(self @ simTrans) for this wave's 16 columns ----
    v8f acc = {};
#pragma unroll
    for (int kt = 0; kt < 4; ++kt) {
        v16bf bfrag = load_b_frag(packS, kt, wave, lane);
        acc = __builtin_amdgcn_wmma_f32_16x16x32_bf16(
            false, a_self[kt], false, bfrag, (short)0, acc, false, false);
    }
#pragma unroll
    for (int r = 0; r < 8; ++r) {
        float v = fmaxf(acc[r], 0.0f);
        simLds[(r + chalf) * 136 + wave * 16 + (lane & 15)] = (__bf16)v;
    }
    __syncthreads();

    v16bf a_sim[4];
#pragma unroll
    for (int kt = 0; kt < 4; ++kt) {
        AFrag f;
        f.h[0] = *(const v8bf*)(&simLds[m * 136 + kt * 32 + khalf]);
        f.h[1] = *(const v8bf*)(&simLds[m * 136 + kt * 32 + 16 + khalf]);
        a_sim[kt] = f.v;
    }

    // ---- out = relu(self@W[0:128] + to@W[128:256] + sim@W[256:384]) ----
    v8f oacc = {};
#pragma unroll
    for (int kt = 0; kt < 4; ++kt) {
        v16bf bfrag = load_b_frag(packW, kt, wave, lane);
        oacc = __builtin_amdgcn_wmma_f32_16x16x32_bf16(
            false, a_self[kt], false, bfrag, (short)0, oacc, false, false);
    }
#pragma unroll
    for (int kt = 0; kt < 4; ++kt) {
        v16bf bfrag = load_b_frag(packW, kt + 4, wave, lane);
        oacc = __builtin_amdgcn_wmma_f32_16x16x32_bf16(
            false, a_to[kt], false, bfrag, (short)0, oacc, false, false);
    }
#pragma unroll
    for (int kt = 0; kt < 4; ++kt) {
        v16bf bfrag = load_b_frag(packW, kt + 8, wave, lane);
        oacc = __builtin_amdgcn_wmma_f32_16x16x32_bf16(
            false, a_sim[kt], false, bfrag, (short)0, oacc, false, false);
    }
#pragma unroll
    for (int r = 0; r < 8; ++r) {
        out[(row0 + r + chalf) * EMB + wave * 16 + (lane & 15)] =
            fmaxf(oacc[r], 0.0f);
    }
}

// ---------------------------------------------------------------------------
extern "C" void kernel_launch(void* const* d_in, const int* in_sizes, int n_in,
                              void* d_out, int out_size, void* d_ws, size_t ws_size,
                              hipStream_t stream) {
    const float* features   = (const float*)d_in[0];
    const float* neigh_mask = (const float*)d_in[1];
    const float* simTrans   = (const float*)d_in[2];
    const float* weight     = (const float*)d_in[3];
    const int*   nodes      = (const int*)d_in[4];
    const int*   neigh_idx  = (const int*)d_in[5];
    float* out = (float*)d_out;

    // workspace layout (bf16 elements)
    __bf16* selfB = (__bf16*)d_ws;                          // B*128
    __bf16* toB   = selfB + (size_t)B_ROWS * FEAT;          // B*128
    __bf16* packS = toB   + (size_t)B_ROWS * FEAT;          // 128*128
    __bf16* packW = packS + 128 * 128;                      // 384*128

    // pack weights into WMMA B-fragment order (tiny, runs each launch)
    pack_b_kernel<<<(4 * 8 * 32 + 255) / 256, 256, 0, stream>>>(simTrans, packS, 128, 128);
    pack_b_kernel<<<(12 * 8 * 32 + 255) / 256, 256, 0, stream>>>(weight, packW, 384, 128);

    // gather + masked neighbor mean (memory-bound, dominant cost)
    agg_kernel<<<B_ROWS / 8, 256, 0, stream>>>(features, neigh_mask, nodes, neigh_idx,
                                               selfB, toB);

    // fused bf16 WMMA GEMMs + relu
    fused_gemm_kernel<<<B_ROWS / 16, 256, 0, stream>>>(selfB, toB, packS, packW, out);
}
